// Gate_47425028882760
// MI455X (gfx1250) — compile-verified, matrix operand order checked by reading, not run on hardware
//
#include <hip/hip_runtime.h>
#include <stdint.h>

#define T_TOKENS   16384
#define NEMBD      7168
#define NEXPERTS   256
#define TOPK       8
#define NGROUPS    8
#define TOPKG      4
#define ROUTE_SCALE 2.5f

#define BM 128
#define BN 128
#define BK 32
#define STR 28    // dword stride per row/col: 112B = 16B aligned; 28r mod 64 has period 16
                  // -> conflict-free b128 fragment reads and staging stores

typedef __attribute__((ext_vector_type(16))) __bf16   v16bf;
typedef __attribute__((ext_vector_type(8)))  float    v8f;
typedef __attribute__((ext_vector_type(8)))  uint32_t v8u;

// Split two consecutive-K fp32 values into WMMA-ready packed bf16 dwords:
// hi = bf16(f1)<<16 | bf16(f0), lo = bf16(f1-hi1)<<16 | bf16(f0-hi0).
__device__ __forceinline__ void pack_pair(float f0, float f1,
                                          uint32_t& hi, uint32_t& lo) {
    uint32_t u0 = __builtin_bit_cast(uint32_t, f0);
    uint32_t u1 = __builtin_bit_cast(uint32_t, f1);
    uint32_t h0 = u0 & 0xFFFF0000u;
    uint32_t h1 = u1 & 0xFFFF0000u;
    float r0 = f0 - __builtin_bit_cast(float, h0);
    float r1 = f1 - __builtin_bit_cast(float, h1);
    hi = h1 | (h0 >> 16);
    lo = ((__builtin_bit_cast(uint32_t, r1) >> 16) << 16) |
         (__builtin_bit_cast(uint32_t, r0) >> 16);
}

// Fragment = 8 dwords already in WMMA 16-bit operand order:
// dword v: K-pair d = (v<4 ? half*4+v : 8+half*4+(v-4)); base points at [row][half*4].
__device__ __forceinline__ v16bf load_frag(const uint32_t* base) {
    const uint4 t0 = *(const uint4*)(base);        // d = h*4 .. h*4+3
    const uint4 t1 = *(const uint4*)(base + 8);    // d = 8+h*4 .. 8+h*4+3
    v8u u;
    u[0] = t0.x; u[1] = t0.y; u[2] = t0.z; u[3] = t0.w;
    u[4] = t1.x; u[5] = t1.y; u[6] = t1.z; u[7] = t1.w;
    return __builtin_bit_cast(v16bf, u);
}

#define WMMA_BF16(A, B, C) \
    __builtin_amdgcn_wmma_f32_16x16x32_bf16(false, (A), false, (B), (short)0, (C), false, false)

// ---------------------------------------------------------------------------
// Kernel 1: scores = sigmoid(X @ W) via bf16 3-term split WMMA, fp32 accum.
// 256 threads / 8 waves; tile BM=128 x BN=128; wave w owns rows [16w,16w+16).
// ---------------------------------------------------------------------------
__global__ __launch_bounds__(256)
void gate_gemm_sigmoid(const float* __restrict__ x,
                       const float* __restrict__ wmat,
                       float* __restrict__ scores) {
    __shared__ __align__(16) uint32_t lds_ah[BM * STR];  // A hi terms, [row][d]
    __shared__ __align__(16) uint32_t lds_al[BM * STR];  // A lo terms
    __shared__ __align__(16) uint32_t lds_bh[BN * STR];  // B hi terms, [col][d]
    __shared__ __align__(16) uint32_t lds_bl[BN * STR];  // B lo terms

    const int tid  = threadIdx.x;
    const int lane = tid & 31;
    const int wv   = tid >> 5;
    const int half = lane >> 4;
    const int l16  = lane & 15;
    const int row0 = blockIdx.x * BM;
    const int n0g  = blockIdx.y * BN;
    const int m0   = wv * 16;

    // Per-thread staging assignment (2 units each for A and B):
    //  A unit: row ar[i] (0..127), k-octet aq[i] (0..3)  -> 8 consecutive K
    //  B unit: k-pair bd[i] (0..15), col-quad bq[i] (0..31)
    int ar[2], aq[2], bd[2], bq[2];
#pragma unroll
    for (int i = 0; i < 2; ++i) {
        int idx = tid + i * 256;       // 0..511
        ar[i] = idx >> 2;  aq[i] = idx & 3;
        bd[i] = idx >> 5;  bq[i] = idx & 31;
    }

    float4 Ar[2][2], Br[2][2];         // register prefetch buffers

    auto load_tile = [&](int kb) {
#pragma unroll
        for (int i = 0; i < 2; ++i) {
            const float* ap = x + (size_t)(row0 + ar[i]) * NEMBD + kb + aq[i] * 8;
            Ar[i][0] = *(const float4*)(ap);
            Ar[i][1] = *(const float4*)(ap + 4);
            const float* bp = wmat + (size_t)(kb + 2 * bd[i]) * NEXPERTS + n0g + bq[i] * 4;
            Br[i][0] = *(const float4*)(bp);             // k = 2d
            Br[i][1] = *(const float4*)(bp + NEXPERTS);  // k = 2d+1
        }
    };

    auto store_tile = [&]() {
#pragma unroll
        for (int i = 0; i < 2; ++i) {
            // A: 4 consecutive K-pairs -> one aligned b128 store per array
            uint4 hv, lv;
            pack_pair(Ar[i][0].x, Ar[i][0].y, hv.x, lv.x);
            pack_pair(Ar[i][0].z, Ar[i][0].w, hv.y, lv.y);
            pack_pair(Ar[i][1].x, Ar[i][1].y, hv.z, lv.z);
            pack_pair(Ar[i][1].z, Ar[i][1].w, hv.w, lv.w);
            *(uint4*)(&lds_ah[ar[i] * STR + aq[i] * 4]) = hv;
            *(uint4*)(&lds_al[ar[i] * STR + aq[i] * 4]) = lv;
            // B: pair across two k-rows, scatter 4 cols (col-major layout)
            uint32_t h, l;
            const int c0 = bq[i] * 4, d = bd[i];
            pack_pair(Br[i][0].x, Br[i][1].x, h, l);
            lds_bh[(c0 + 0) * STR + d] = h; lds_bl[(c0 + 0) * STR + d] = l;
            pack_pair(Br[i][0].y, Br[i][1].y, h, l);
            lds_bh[(c0 + 1) * STR + d] = h; lds_bl[(c0 + 1) * STR + d] = l;
            pack_pair(Br[i][0].z, Br[i][1].z, h, l);
            lds_bh[(c0 + 2) * STR + d] = h; lds_bl[(c0 + 2) * STR + d] = l;
            pack_pair(Br[i][0].w, Br[i][1].w, h, l);
            lds_bh[(c0 + 3) * STR + d] = h; lds_bl[(c0 + 3) * STR + d] = l;
        }
    };

    v8f acc[8];
#pragma unroll
    for (int i = 0; i < 8; ++i) acc[i] = (v8f)0.0f;

    load_tile(0);
    for (int kb = 0; kb < NEMBD; kb += BK) {
        store_tile();
        __syncthreads();
        if (kb + BK < NEMBD) load_tile(kb + BK);   // prefetch next tile (overlaps WMMAs)

        const int abase = (m0 + l16) * STR + half * 4;
        const v16bf ahi = load_frag(&lds_ah[abase]);
        const v16bf alo = load_frag(&lds_al[abase]);

#pragma unroll
        for (int p = 0; p < 4; ++p) {              // ct pairs: interleave accs -> no D-hazard chains
            const int c0 = (2 * p) * 16 + l16;
            const int c1 = (2 * p + 1) * 16 + l16;
            const v16bf bh0 = load_frag(&lds_bh[c0 * STR + half * 4]);
            const v16bf bl0 = load_frag(&lds_bl[c0 * STR + half * 4]);
            const v16bf bh1 = load_frag(&lds_bh[c1 * STR + half * 4]);
            const v16bf bl1 = load_frag(&lds_bl[c1 * STR + half * 4]);
            acc[2 * p]     = WMMA_BF16(ahi, bh0, acc[2 * p]);
            acc[2 * p + 1] = WMMA_BF16(ahi, bh1, acc[2 * p + 1]);
            acc[2 * p]     = WMMA_BF16(ahi, bl0, acc[2 * p]);
            acc[2 * p + 1] = WMMA_BF16(ahi, bl1, acc[2 * p + 1]);
            acc[2 * p]     = WMMA_BF16(alo, bh0, acc[2 * p]);
            acc[2 * p + 1] = WMMA_BF16(alo, bh1, acc[2 * p + 1]);
        }
        __syncthreads();
    }

    // epilogue: sigmoid, write original_scores
#pragma unroll
    for (int ct = 0; ct < 8; ++ct) {
        const int col = n0g + ct * 16 + l16;
#pragma unroll
        for (int r = 0; r < 8; ++r) {
            const int row = row0 + m0 + half * 8 + r;   // C layout: M = 8*half + vgpr
            const float s = acc[ct][r];
            scores[(size_t)row * NEXPERTS + col] = 1.0f / (1.0f + __expf(-s));
        }
    }
}

// ---------------------------------------------------------------------------
// Kernel 2: group-limited routing. One wave32 per token; lane l owns experts
// [8l, 8l+8) so each 32-expert group == exactly 4 lanes.
// ---------------------------------------------------------------------------
__global__ __launch_bounds__(256)
void gate_route_topk(const float* __restrict__ scores,
                     const float* __restrict__ bias,
                     float* __restrict__ w_out,
                     int* __restrict__ idx_out) {
    const int lane = threadIdx.x & 31;
    const int wv   = threadIdx.x >> 5;
    const int t    = blockIdx.x * 8 + wv;
    const float* srow = scores + (size_t)t * NEXPERTS;

    float s[8], sb[8];
    const float4 v0 = *(const float4*)(srow + lane * 8);
    const float4 v1 = *(const float4*)(srow + lane * 8 + 4);
    s[0] = v0.x; s[1] = v0.y; s[2] = v0.z; s[3] = v0.w;
    s[4] = v1.x; s[5] = v1.y; s[6] = v1.z; s[7] = v1.w;
#pragma unroll
    for (int i = 0; i < 8; ++i) sb[i] = s[i] + bias[lane * 8 + i];

    const float NEG = -__builtin_inff();

    // lane-local top-2 of 8 biased scores
    float t1 = NEG, t2 = NEG;
#pragma unroll
    for (int i = 0; i < 8; ++i) {
        float v = sb[i];
        if (v > t1) { t2 = t1; t1 = v; } else if (v > t2) { t2 = v; }
    }
    // merge top-2 across the 4 lanes of each group (xor 1, xor 2 stay in-group)
#pragma unroll
    for (int m = 1; m <= 2; m <<= 1) {
        float o1 = __shfl_xor(t1, m);
        float o2 = __shfl_xor(t2, m);
        float n1 = fmaxf(t1, o1);
        float n2 = fmaxf(fminf(t1, o1), fmaxf(t2, o2));
        t1 = n1; t2 = n2;
    }
    const float gs = t1 + t2;                     // group score (uniform per group)

    float gall[8];
#pragma unroll
    for (int g = 0; g < 8; ++g) gall[g] = __shfl(gs, g * 4);

    // top-4 groups (stable: strict > picks lowest index on ties, like jax)
    unsigned gsel = 0;
#pragma unroll
    for (int it = 0; it < TOPKG; ++it) {
        float best = NEG; int bi = 0;
#pragma unroll
        for (int g = 0; g < 8; ++g)
            if (!((gsel >> g) & 1u) && gall[g] > best) { best = gall[g]; bi = g; }
        gsel |= 1u << bi;
    }
    const bool keep = (gsel >> (lane >> 2)) & 1u;

    float avail[8];
#pragma unroll
    for (int i = 0; i < 8; ++i) avail[i] = keep ? sb[i] : 0.0f;

    float myw = 0.0f; int myi = 0; float wsum = 0.0f;
#pragma unroll
    for (int k = 0; k < TOPK; ++k) {
        // lane-local argmax over remaining
        float lb = NEG; int li = 0;
#pragma unroll
        for (int i = 0; i < 8; ++i)
            if (avail[i] > lb) { lb = avail[i]; li = i; }
        float bv = lb; int bidx = lane * 8 + li; float bsv = s[li];
        // wave-wide argmax (butterfly); ties -> lower expert index (stable)
#pragma unroll
        for (int m = 16; m >= 1; m >>= 1) {
            float ov = __shfl_xor(bv, m);
            int   oi = __shfl_xor(bidx, m);
            float os = __shfl_xor(bsv, m);
            if (ov > bv || (ov == bv && oi < bidx)) { bv = ov; bidx = oi; bsv = os; }
        }
        if ((bidx >> 3) == lane) avail[bidx & 7] = NEG;  // owner removes winner
        wsum += bsv;                                      // unbiased sigmoid score
        if (lane == k) { myw = bsv; myi = bidx; }
    }
    const float scale = ROUTE_SCALE / (wsum + 1e-6f);
    if (lane < TOPK) {
        w_out[(size_t)t * TOPK + lane]  = myw * scale;
        idx_out[(size_t)t * TOPK + lane] = myi;
    }
}

extern "C" void kernel_launch(void* const* d_in, const int* in_sizes, int n_in,
                              void* d_out, int out_size, void* d_ws, size_t ws_size,
                              hipStream_t stream) {
    (void)in_sizes; (void)n_in; (void)out_size; (void)d_ws; (void)ws_size;
    const float* x    = (const float*)d_in[0];
    const float* wmat = (const float*)d_in[1];
    const float* bias = (const float*)d_in[2];

    float* scores = (float*)d_out;                                  // [T, 256]
    float* w_out  = scores + (size_t)T_TOKENS * NEXPERTS;           // [T, 8]
    int*   i_out  = (int*)(w_out + (size_t)T_TOKENS * TOPK);        // [T, 8] int32

    dim3 g1(T_TOKENS / BM, NEXPERTS / BN);   // 128 x 2 workgroups
    gate_gemm_sigmoid<<<g1, 256, 0, stream>>>(x, wmat, scores);
    gate_route_topk<<<T_TOKENS / 8, 256, 0, stream>>>(scores, bias, w_out, i_out);
}
